// DPSOG_Water_309237645340
// MI455X (gfx1250) — matrix operand order repaired and checked by reading, not compiled, not run on hardware
//
#include <hip/hip_runtime.h>

#define DEVFN __device__ __forceinline__

typedef __attribute__((ext_vector_type(2)))  float        v2f;
typedef __attribute__((ext_vector_type(8)))  float        v8f;
typedef __attribute__((ext_vector_type(16))) _Float16     v16h;
typedef __attribute__((ext_vector_type(4)))  unsigned int v4u;
typedef __attribute__((ext_vector_type(8)))  int          v8i;
typedef __attribute__((ext_vector_type(4)))  int          v4i;

namespace cfg {
constexpr float L = 14.4088f;
constexpr int   B = 8, N_O = 256, N_H = 512, NPT = 768;
constexpr float RIN_A = 2.f, RC_A = 4.f, RIN_R = 3.f, RC_R = 6.f;
constexpr int   NK = 500, NKP = 512;        // padded k stride
constexpr float NORM_A = 1764.f;            // (14+28)^2
constexpr int   MA = 42, MR = 180;          // neighbors per atom (angular/radial)
constexpr int   DDIM = 544;                 // 16*32 + 32

// flattened input indices (setup_inputs dict order, params nested in insertion order)
constexpr int IN_X = 0, IN_IOOA = 1, IN_IOHA = 2, IN_IHOA = 3, IN_IHHA = 4,
              IN_IOOR = 5, IN_IOHR = 6, IN_IHOR = 7, IN_IHHR = 8,
              IN_PYR = 10,      // 8 pyramids x (5 layers x {W,b}) = 80 arrays
              IN_PYRQ = 90,     // 3 layers x {W,b}
              IN_FIT = 96,      // 5 layers x {W,b}
              IN_LIN = 106, IN_WQ = 108, IN_SIGMA = 109, IN_AMP = 110, IN_KVEC = 111;

// workspace offsets in floats
constexpr size_t WS_SA = 0;
constexpr size_t WS_RA = WS_SA + (size_t)B * NPT * MA;
constexpr size_t WS_GA = WS_RA + (size_t)B * NPT * MA * 4;
constexpr size_t WS_DR = WS_GA + (size_t)B * NPT * MA * 32;
constexpr size_t WS_D  = WS_DR + (size_t)B * NPT * 32;
constexpr size_t WS_Q  = WS_D  + (size_t)B * NPT * DDIM;
constexpr size_t WS_RE = WS_Q  + (size_t)B * NPT * 4;
constexpr size_t WS_IM = WS_RE + (size_t)B * 4 * NKP;
} // namespace cfg

struct PtrPack { const void* p[112]; };

// ---------------------------------------------------------------------------
// helpers
// ---------------------------------------------------------------------------
DEVFN float pair_s(const float* xi, const float* xj, float rin, float rc,
                   float d[3], float* rout) {
  float r2 = 1e-12f;
#pragma unroll
  for (int k = 0; k < 3; ++k) {
    float dd = xj[k] - xi[k];
    dd -= cfg::L * rintf(dd / cfg::L);
    d[k] = dd;
    r2 = fmaf(dd, dd, r2);
  }
  float r = sqrtf(r2);
  float u = fminf(fmaxf((r - rin) / (rc - rin), 0.f), 1.f);
  float sw = ((-6.f * u + 15.f) * u - 10.f) * u * u * u + 1.f;
  *rout = r;
  return sw / r;
}

// descriptor pyramid: dims 1->2->4->8->16->32, every layer is the doubling
// branch: x_new[j] = x_old[j % din] + tanh((xW+b)[j])
DEVFN void pyr_desc(float s0, const PtrPack& P, int base, float* out) {
  float x[32];
  x[0] = s0;
  int din = 1;
#pragma unroll
  for (int l = 0; l < 5; ++l) {
    const int dout = 2 << l;  // 2,4,8,16,32
    const float* W  = (const float*)P.p[base + 2 * l];
    const float* bb = (const float*)P.p[base + 2 * l + 1];
    float y[32];
    for (int j = 0; j < dout; ++j) {
      float acc = bb[j];
      for (int i = 0; i < din; ++i) acc = fmaf(x[i], W[i * dout + j], acc);
      y[j] = x[j & (din - 1)] + tanhf(acc);
    }
    for (int j = 0; j < dout; ++j) x[j] = y[j];
    din = dout;
  }
#pragma unroll
  for (int j = 0; j < 32; ++j) out[j] = x[j];
}

// Tensor Data Mover: linear copy of nelem f32 from global to LDS.
// D# group0: count=1, lds_addr, global_addr, type=2.
// D# group1: data_size=4B, tensor_dim0=nelem, tensor_dim1=1, tile_dim0=nelem,
//            tensor_dim0_stride=nelem.
// This toolchain exposes the 6-arg builtin: (v4u, v8i, v4i, v4i, v8i, cpol).
DEVFN void tdm_load_f32(const float* src, unsigned lds_off, unsigned nelem) {
  unsigned long long ga = (unsigned long long)(size_t)src;
  v4u g0;
  g0.x = 1u;                                            // count=1
  g0.y = lds_off;                                       // lds_addr
  g0.z = (unsigned)(ga & 0xFFFFFFFFull);                // global_addr[31:0]
  g0.w = (unsigned)(((ga >> 32) & 0x01FFFFFFull) | (2u << 30));  // addr hi | type=2
  v8i g1;
  g1[0] = (int)(2u << 16);                              // data_size = 4 bytes
  g1[1] = (int)((nelem & 0xFFFFu) << 16);               // tensor_dim0[15:0]
  g1[2] = (int)(((nelem >> 16) & 0xFFFFu) | (1u << 16));// dim0 hi | tensor_dim1=1
  g1[3] = (int)((nelem & 0xFFFFu) << 16);               // tile_dim0
  g1[4] = 0;                                            // tile_dim1/2 unused
  g1[5] = (int)nelem;                                   // tensor_dim0_stride lo
  g1[6] = 0;
  g1[7] = 0;
  v4i z4 = {0, 0, 0, 0};
  v8i z8 = {0, 0, 0, 0, 0, 0, 0, 0};
  __builtin_amdgcn_tensor_load_to_lds(g0, g1, z4, z4, z8, 0);
}

// ---------------------------------------------------------------------------
// kernels
// ---------------------------------------------------------------------------
__global__ void k_zero(float* out, int n) {
  int t = blockIdx.x * blockDim.x + threadIdx.x;
  if (t < n) out[t] = 0.f;
}

// angular pair features: s (b,n,42) and R (b,n,42,4)
__global__ void k_pair_ang(PtrPack P, float* ws) {
  int t = blockIdx.x * blockDim.x + threadIdx.x;
  if (t >= cfg::B * cfg::NPT * cfg::MA) return;
  int m = t % cfg::MA;
  int n = (t / cfg::MA) % cfg::NPT;
  int b = t / (cfg::MA * cfg::NPT);
  const float* X = (const float*)P.p[cfg::IN_X];
  int nbr;
  if (n < cfg::N_O) {
    if (m < 14) nbr = ((const int*)P.p[cfg::IN_IOOA])[((size_t)b * cfg::N_O + n) * 14 + m];
    else        nbr = cfg::N_O + ((const int*)P.p[cfg::IN_IOHA])[((size_t)b * cfg::N_O + n) * 28 + (m - 14)];
  } else {
    int h = n - cfg::N_O;
    if (m < 14) nbr = ((const int*)P.p[cfg::IN_IHOA])[((size_t)b * cfg::N_H + h) * 14 + m];
    else        nbr = cfg::N_O + ((const int*)P.p[cfg::IN_IHHA])[((size_t)b * cfg::N_H + h) * 28 + (m - 14)];
  }
  const float* xi = X + ((size_t)b * cfg::NPT + n) * 3;
  const float* xj = X + ((size_t)b * cfg::NPT + nbr) * 3;
  float d[3], r;
  float s = pair_s(xi, xj, cfg::RIN_A, cfg::RC_A, d, &r);
  ws[cfg::WS_SA + t] = s;
  float* R = ws + cfg::WS_RA + (size_t)t * 4;
  R[0] = s;
#pragma unroll
  for (int k = 0; k < 3; ++k) R[k + 1] = s * d[k] / r;
}

// angular G: per (b,n,m) run species-pair pyramid -> 32 features
__global__ void k_G_ang(PtrPack P, float* ws) {
  int t = blockIdx.x * blockDim.x + threadIdx.x;
  if (t >= cfg::B * cfg::NPT * cfg::MA) return;
  int m = t % cfg::MA;
  int n = (t / cfg::MA) % cfg::NPT;
  int p = (n < cfg::N_O) ? ((m < 14) ? 0 : 1) : ((m < 14) ? 2 : 3);
  float g[32];
  pyr_desc(ws[cfg::WS_SA + t], P, cfg::IN_PYR + p * 10, g);
  float* G = ws + cfg::WS_GA + (size_t)t * 32;
#pragma unroll
  for (int c = 0; c < 32; ++c) G[c] = g[c];
}

// radial: wave per atom, lane per neighbor chunk; LDS cross-lane reduce -> Dr mean
__global__ void __launch_bounds__(256) k_radial(PtrPack P, float* ws) {
  __shared__ float red[8][1024];
  int wave = threadIdx.x >> 5, lane = threadIdx.x & 31;
  int atom = blockIdx.x * 8 + wave;
  int b = atom / cfg::NPT, n = atom % cfg::NPT;
  const float* X = (const float*)P.p[cfg::IN_X];
  const float* xi = X + ((size_t)b * cfg::NPT + n) * 3;
  float acc[32];
#pragma unroll
  for (int c = 0; c < 32; ++c) acc[c] = 0.f;
  for (int it = 0; it < 6; ++it) {
    int m = it * 32 + lane;
    if (m < cfg::MR) {
      int nbr, p;
      if (n < cfg::N_O) {
        if (m < 60) { nbr = ((const int*)P.p[cfg::IN_IOOR])[((size_t)b * cfg::N_O + n) * 60 + m]; p = 4; }
        else        { nbr = cfg::N_O + ((const int*)P.p[cfg::IN_IOHR])[((size_t)b * cfg::N_O + n) * 120 + (m - 60)]; p = 5; }
      } else {
        int h = n - cfg::N_O;
        if (m < 60) { nbr = ((const int*)P.p[cfg::IN_IHOR])[((size_t)b * cfg::N_H + h) * 60 + m]; p = 6; }
        else        { nbr = cfg::N_O + ((const int*)P.p[cfg::IN_IHHR])[((size_t)b * cfg::N_H + h) * 120 + (m - 60)]; p = 7; }
      }
      float d[3], r;
      float s = pair_s(xi, X + ((size_t)b * cfg::NPT + nbr) * 3, cfg::RIN_R, cfg::RC_R, d, &r);
      float g[32];
      pyr_desc(s, P, cfg::IN_PYR + p * 10, g);
#pragma unroll
      for (int c = 0; c < 32; ++c) acc[c] += g[c];
    }
  }
#pragma unroll
  for (int c = 0; c < 32; ++c) red[wave][lane * 32 + c] = acc[c];
  __syncthreads();
  float tsum = 0.f;
  for (int l = 0; l < 32; ++l) tsum += red[wave][l * 32 + lane];
  ws[cfg::WS_DR + (size_t)atom * 32 + lane] = tsum / (float)cfg::MR;
}

// descriptor: T = R^T G (4x32, lane-resident columns), then
// Da = T[:, :16]^T x T via two V_WMMA_F32_16X16X4_F32 (M=16,N=16,K=4, f32)
__global__ void __launch_bounds__(256) k_desc(PtrPack P, float* ws) {
  int wave = threadIdx.x >> 5, lane = threadIdx.x & 31;
  int atom = blockIdx.x * 8 + wave;
  const float* Rr = ws + cfg::WS_RA + (size_t)atom * cfg::MA * 4;
  const float* Gr = ws + cfg::WS_GA + (size_t)atom * cfg::MA * 32;
  float t0 = 0.f, t1 = 0.f, t2 = 0.f, t3 = 0.f;  // T[a][lane]
  for (int m = 0; m < cfg::MA; ++m) {
    float g = Gr[m * 32 + lane];
    t0 = fmaf(Rr[m * 4 + 0], g, t0);
    t1 = fmaf(Rr[m * 4 + 1], g, t1);
    t2 = fmaf(Rr[m * 4 + 2], g, t2);
    t3 = fmaf(Rr[m * 4 + 3], g, t3);
  }
  bool lo = lane < 16;
  int l15 = lane & 15;
  // A (16x4): lanes0-15 hold (A[l][0],A[l][1]); lanes16-31 hold (A[l][2],A[l][3])
  float t2s = __shfl(t2, l15, 32), t3s = __shfl(t3, l15, 32);
  v2f a; a.x = lo ? t0 : t2s; a.y = lo ? t1 : t3s;
  // B tile covering N=0..15 equals A; tile N=16..31 gathers T columns 16..31
  float s0 = __shfl(t0, l15 + 16, 32), s1 = __shfl(t1, l15 + 16, 32);
  float s2 = __shfl(t2, l15 + 16, 32), s3 = __shfl(t3, l15 + 16, 32);
  v2f b1; b1.x = lo ? s0 : s2; b1.y = lo ? s1 : s3;
  v8f c0 = {}; v8f c1 = {};
  c0 = __builtin_amdgcn_wmma_f32_16x16x4_f32(false, a, false, a,  (short)0, c0, false, false);
  // identical opcode as previous, shared A operand -> reuse-A hint
  c1 = __builtin_amdgcn_wmma_f32_16x16x4_f32(false, a, false, b1, (short)0, c1, true, false);
  float* D = ws + cfg::WS_D + (size_t)atom * cfg::DDIM;
  const float inv = 1.f / cfg::NORM_A;
#pragma unroll
  for (int r = 0; r < 8; ++r) {
    int srow = r + (lo ? 0 : 8);
    D[srow * 32 + l15]      = c0[r] * inv;
    D[srow * 32 + 16 + l15] = c1[r] * inv;
  }
  D[512 + lane] = ws[cfg::WS_DR + (size_t)atom * 32 + lane];
}

// fit + Q heads: wave per 16-atom tile; layer-1 (16x544x16) via f16 WMMA with
// TDM-staged weights in LDS; tiny tail layers scalar per atom.
__global__ void __launch_bounds__(32) k_fit(PtrPack P, float* ws, float* out) {
  __shared__ float w1f[cfg::DDIM * 16];
  __shared__ float w1q[cfg::DDIM * 16];
  __shared__ float actF[16][17];
  __shared__ float actQ[16][17];
  int lane = threadIdx.x & 31;
  int atom0 = blockIdx.x * 16;
  int b = atom0 / cfg::NPT;

  tdm_load_f32((const float*)P.p[cfg::IN_FIT + 0],  (unsigned)(size_t)&w1f[0], cfg::DDIM * 16);
  tdm_load_f32((const float*)P.p[cfg::IN_PYRQ + 0], (unsigned)(size_t)&w1q[0], cfg::DDIM * 16);
  __builtin_amdgcn_s_wait_tensorcnt(0);
  __syncthreads();

  bool lo = lane < 16;
  int row = lane & 15;                         // atom within tile / B column
  int khi = lo ? 0 : 8;                        // unused
  int krow = lo ? 0 : 16;                      // B K-sublane offset
  const float* Drow = ws + cfg::WS_D + (size_t)(atom0 + row) * cfg::DDIM;
  __builtin_prefetch(Drow, 0, 1);
  v8f cf = {}; v8f cq = {};
#pragma unroll
  for (int t = 0; t < 17; ++t) {               // fully unrolled: 34 WMMAs in flight
    int k0 = t * 32;
    v16h av, bf, bq;
#pragma unroll
    for (int j = 0; j < 16; ++j) {             // A 16-bit 16x32 layout
      av[j]     = (_Float16)Drow[k0 + krow + j];
    }
#pragma unroll
    for (int j = 0; j < 16; ++j) {             // B 16-bit 32x16 layout
      bf[j] = (_Float16)w1f[(k0 + krow + j) * 16 + row];
      bq[j] = (_Float16)w1q[(k0 + krow + j) * 16 + row];
    }
    cf = __builtin_amdgcn_wmma_f32_16x16x32_f16(false, av, false, bf, (short)0, cf, false, false);
    // identical opcode as previous, shared A operand -> reuse-A hint
    cq = __builtin_amdgcn_wmma_f32_16x16x32_f16(false, av, false, bq, (short)0, cq, true, false);
  }
  const float* b1f = (const float*)P.p[cfg::IN_FIT + 1];
  const float* b1q = (const float*)P.p[cfg::IN_PYRQ + 1];
  int aro = lo ? 0 : 8;
#pragma unroll
  for (int r = 0; r < 8; ++r) {
    actF[r + aro][row] = tanhf(cf[r] + b1f[row]);
    actQ[r + aro][row] = tanhf(cq[r] + b1q[row]);
  }
  __syncthreads();

  if (lane < 16) {
    // fit tail: 16->8->4->2->1, plain tanh layers
    float x[16], y[16];
#pragma unroll
    for (int i = 0; i < 16; ++i) x[i] = actF[lane][i];
    int din = 16;
#pragma unroll
    for (int l = 1; l < 5; ++l) {
      int dout = 16 >> l;  // 8,4,2,1
      const float* W  = (const float*)P.p[cfg::IN_FIT + 2 * l];
      const float* bb = (const float*)P.p[cfg::IN_FIT + 2 * l + 1];
      for (int j = 0; j < dout; ++j) {
        float accv = bb[j];
        for (int i = 0; i < din; ++i) accv = fmaf(x[i], W[i * dout + j], accv);
        y[j] = tanhf(accv);
      }
      for (int j = 0; j < dout; ++j) x[j] = y[j];
      din = dout;
    }
    float Wl = ((const float*)P.p[cfg::IN_LIN])[0];
    float bl = ((const float*)P.p[cfg::IN_LIN + 1])[0];
    atomicAdd(&out[b], x[0] * Wl + bl);

    // Q tail: two residual 16->16 layers, then q = Q @ Wq
    float qx[16];
#pragma unroll
    for (int i = 0; i < 16; ++i) qx[i] = actQ[lane][i];
#pragma unroll
    for (int l = 1; l < 3; ++l) {
      const float* W  = (const float*)P.p[cfg::IN_PYRQ + 2 * l];
      const float* bb = (const float*)P.p[cfg::IN_PYRQ + 2 * l + 1];
      float qy[16];
      for (int j = 0; j < 16; ++j) {
        float accv = bb[j];
        for (int i = 0; i < 16; ++i) accv = fmaf(qx[i], W[i * 16 + j], accv);
        qy[j] = qx[j] + tanhf(accv);
      }
      for (int j = 0; j < 16; ++j) qx[j] = qy[j];
    }
    const float* Wq = (const float*)P.p[cfg::IN_WQ];
    float* qout = ws + cfg::WS_Q + (size_t)(atom0 + lane) * 4;
#pragma unroll
    for (int c = 0; c < 4; ++c) {
      float accv = 0.f;
      for (int i = 0; i < 16; ++i) accv = fmaf(qx[i], Wq[i * 4 + c], accv);
      qout[c] = accv;
    }
  }
}

// Ewald structure factors: S[k,c] = sum_n trig(x_n . k) q[n,c]
// wave per (batch, 16-k tile); K over atoms in chunks of 4 (f32 WMMA)
__global__ void __launch_bounds__(32) k_ewald_s(PtrPack P, float* ws) {
  int lane = threadIdx.x & 31;
  int tile = blockIdx.x & 31;
  int b = blockIdx.x >> 5;
  bool lo = lane < 16;
  int l15 = lane & 15;
  int kk = tile * 16 + l15;
  int kc = kk < cfg::NK ? kk : cfg::NK - 1;
  const float* kv = (const float*)P.p[cfg::IN_KVEC] + 3 * kc;
  float kx = kv[0], ky = kv[1], kz = kv[2];
  const float* X  = (const float*)P.p[cfg::IN_X] + (size_t)b * cfg::NPT * 3;
  const float* qw = ws + cfg::WS_Q + (size_t)b * cfg::NPT * 4;
  v8f cre = {}; v8f cim = {};
  int c = l15;  // channel (N dim), only c<4 valid
#pragma unroll 4
  for (int n0 = 0; n0 < cfg::NPT; n0 += 4) {
    int na = n0 + (lo ? 0 : 2);
    const float* xa = X + (size_t)na * 3;
    float thA = xa[0] * kx + xa[1] * ky + xa[2] * kz;
    float thB = xa[3] * kx + xa[4] * ky + xa[5] * kz;
    v2f ac; ac.x = cosf(thA); ac.y = cosf(thB);
    v2f as; as.x = sinf(thA); as.y = sinf(thB);
    float q0 = 0.f, q1 = 0.f;
    if (c < 4) {
      q0 = qw[(size_t)na * 4 + c];
      q1 = qw[(size_t)(na + 1) * 4 + c];
    }
    v2f bq; bq.x = q0; bq.y = q1;
    cre = __builtin_amdgcn_wmma_f32_16x16x4_f32(false, ac, false, bq, (short)0, cre, false, false);
    // identical opcode as previous, shared B operand -> reuse-B hint
    cim = __builtin_amdgcn_wmma_f32_16x16x4_f32(false, as, false, bq, (short)0, cim, false, true);
  }
  if (c < 4) {
#pragma unroll
    for (int r = 0; r < 8; ++r) {
      int k = tile * 16 + r + (lo ? 0 : 8);
      if (k < cfg::NK) {
        ws[cfg::WS_RE + ((size_t)b * 4 + c) * cfg::NKP + k] = cre[r];
        ws[cfg::WS_IM + ((size_t)b * 4 + c) * cfg::NKP + k] = cim[r];
      }
    }
  }
}

// per-atom long-range energy + analytic phase-gradient force contribution
__global__ void k_ewald_ef(PtrPack P, float* ws, float* out) {
  int t = blockIdx.x * blockDim.x + threadIdx.x;
  if (t >= cfg::B * cfg::NPT) return;
  int b = t / cfg::NPT;
  const float* Xp = (const float*)P.p[cfg::IN_X] + (size_t)t * 3;
  float px = Xp[0], py = Xp[1], pz = Xp[2];
  const float* KV  = (const float*)P.p[cfg::IN_KVEC];
  const float* amp = (const float*)P.p[cfg::IN_AMP];
  const float* sig = (const float*)P.p[cfg::IN_SIGMA];
  float a2[4], s2[4], qn[4];
#pragma unroll
  for (int c = 0; c < 4; ++c) {
    a2[c] = amp[c] * amp[c];
    s2[c] = sig[c] * sig[c];
    qn[c] = ws[cfg::WS_Q + (size_t)t * 4 + c];
  }
  const float* Re = ws + cfg::WS_RE + (size_t)b * 4 * cfg::NKP;
  const float* Im = ws + cfg::WS_IM + (size_t)b * 4 * cfg::NKP;
  float esum = 0.f, fx = 0.f, fy = 0.f, fz = 0.f;
  for (int k = 0; k < cfg::NK; ++k) {
    float kx = KV[3 * k], ky = KV[3 * k + 1], kz = KV[3 * k + 2];
    float k2 = kx * kx + ky * ky + kz * kz;
    float th = px * kx + py * ky + pz * kz;
    float cth = cosf(th), sth = sinf(th);
    float g = 0.f;
#pragma unroll
    for (int c = 0; c < 4; ++c) {
      float w  = a2[c] * expf(-0.5f * s2[c] * k2) / k2;
      float re = Re[c * cfg::NKP + k], im = Im[c * cfg::NKP + k];
      esum = fmaf(qn[c] * w, cth * re + sth * im, esum);
      g    = fmaf(qn[c] * w, im * cth - re * sth, g);
    }
    fx = fmaf(g, kx, fx);
    fy = fmaf(g, ky, fy);
    fz = fmaf(g, kz, fz);
  }
  const float invL3 = 1.f / (cfg::L * cfg::L * cfg::L);
  atomicAdd(&out[b], 0.5f * invL3 * esum);
  float* F = out + cfg::B + (size_t)t * 3;
  F[0] = -invL3 * fx;
  F[1] = -invL3 * fy;
  F[2] = -invL3 * fz;
}

// ---------------------------------------------------------------------------
extern "C" void kernel_launch(void* const* d_in, const int* in_sizes, int n_in,
                              void* d_out, int out_size, void* d_ws, size_t ws_size,
                              hipStream_t stream) {
  (void)in_sizes; (void)ws_size;
  PtrPack P{};
  for (int i = 0; i < n_in && i < 112; ++i) P.p[i] = d_in[i];
  float* ws  = (float*)d_ws;
  float* out = (float*)d_out;

  k_zero<<<(out_size + 255) / 256, 256, 0, stream>>>(out, out_size);
  const int nPA = cfg::B * cfg::NPT * cfg::MA;
  k_pair_ang<<<(nPA + 255) / 256, 256, 0, stream>>>(P, ws);
  k_G_ang   <<<(nPA + 255) / 256, 256, 0, stream>>>(P, ws);
  k_radial  <<<cfg::B * cfg::NPT / 8, 256, 0, stream>>>(P, ws);
  k_desc    <<<cfg::B * cfg::NPT / 8, 256, 0, stream>>>(P, ws);
  k_fit     <<<cfg::B * cfg::NPT / 16, 32, 0, stream>>>(P, ws, out);
  k_ewald_s <<<cfg::B * 32, 32, 0, stream>>>(P, ws);
  k_ewald_ef<<<(cfg::B * cfg::NPT + 255) / 256, 256, 0, stream>>>(P, ws, out);
}
